// NonLocalBlock_31464930411027
// MI455X (gfx1250) — matrix operand order repaired and checked by reading, not compiled, not run on hardware
//
#include <hip/hip_runtime.h>

typedef __attribute__((ext_vector_type(16))) _Float16 v16h;
typedef __attribute__((ext_vector_type(8)))  _Float16 v8h;
typedef __attribute__((ext_vector_type(8)))  float    v8f;

#define NWIN  2048
#define NH    16
#define HD    8
#define INTER 128
#define CIN   256
#define NTOK  65
#define IMG   128
#define QSCALE 0.35355339059327373f  // 1/sqrt(8)

static __device__ __forceinline__ v8f wmma_f16(v16h a, v16h b, v8f c) {
  // D = A(16x32 f16) * B(32x16 f16) + C(16x16 f32)
  return __builtin_amdgcn_wmma_f32_16x16x32_f16(false, a, false, b, (short)0, c,
                                                false, false);
}

// ---------------------------------------------------------------------------
// Stage 1: QKV projection. One block per 8x8 window. GEMM M=128,N=64,K=256.
// LDS holds x tiles transposed [pixel][channel] (k-contiguous for B frags).
// ---------------------------------------------------------------------------
__global__ __launch_bounds__(256) void k_qkv(
    const float* __restrict__ xthis, const float* __restrict__ xother,
    const float* __restrict__ Wq, const float* __restrict__ bq,
    const float* __restrict__ Wk, const float* __restrict__ bk,
    const float* __restrict__ Wv, const float* __restrict__ bv,
    const float* __restrict__ gtok,
    _Float16* __restrict__ Qws, _Float16* __restrict__ Kws,
    _Float16* __restrict__ Vws) {
  __shared__ __align__(16) _Float16 Xo[64][264];
  __shared__ __align__(16) _Float16 Xt[64][264];
  const int w = blockIdx.x;
  const int b = w >> 8, wh = (w >> 4) & 15, ww = w & 15;
  const int tid = threadIdx.x;

  for (int idx = tid; idx < CIN * 64; idx += 256) {
    const int c = idx >> 6, n = idx & 63;
    const int hp = wh * 8 + (n >> 3), wp = ww * 8 + (n & 7);
    const int gi = ((b * CIN + c) * IMG + hp) * IMG + wp;
    Xo[n][c] = (_Float16)xother[gi];
    Xt[n][c] = (_Float16)xthis[gi];
  }
  // Global token -> token slot 0 of every window (Q gets the 1/sqrt(hd) scale).
  if (tid < INTER) {
    const float g = gtok[tid];
    const int base = ((w * NH + (tid >> 3)) * NTOK) * HD + (tid & 7);
    Qws[base] = (_Float16)(g * QSCALE);
    Kws[base] = (_Float16)g;
    Vws[base] = (_Float16)g;
  }
  __syncthreads();

  const int lane = tid & 31, hlf = lane >> 4, lm = lane & 15;
  const int mtile = tid >> 5;  // 8 waves -> 8 tiles of 16 output channels

  for (int mat = 0; mat < 3; ++mat) {
    const float* Wm = (mat == 0) ? Wq : (mat == 1) ? Wk : Wv;
    const float* bm = (mat == 0) ? bq : (mat == 1) ? bk : bv;
    _Float16* Ows = (mat == 0) ? Qws : (mat == 1) ? Kws : Vws;
    const _Float16(*X)[264] = (mat == 0) ? Xo : Xt;

    v8f acc[4] = {};
    for (int ks = 0; ks < 8; ++ks) {
      // A fragment: row m = mtile*16+lm, k runs of 8 split by half-wave.
      v16h a;
      {
        const int m = mtile * 16 + lm;
        const float* wr = Wm + m * CIN + ks * 32 + hlf * 8;
#pragma unroll
        for (int g = 0; g < 2; ++g) {
          const float4 f0 = *(const float4*)(wr + g * 16);
          const float4 f1 = *(const float4*)(wr + g * 16 + 4);
          a[g * 8 + 0] = (_Float16)f0.x; a[g * 8 + 1] = (_Float16)f0.y;
          a[g * 8 + 2] = (_Float16)f0.z; a[g * 8 + 3] = (_Float16)f0.w;
          a[g * 8 + 4] = (_Float16)f1.x; a[g * 8 + 5] = (_Float16)f1.y;
          a[g * 8 + 6] = (_Float16)f1.z; a[g * 8 + 7] = (_Float16)f1.w;
        }
      }
#pragma unroll
      for (int nt = 0; nt < 4; ++nt) {
        // B fragment: col n = pixel, k runs of 8 split by half-wave.
        const _Float16* xr = &X[nt * 16 + lm][ks * 32 + hlf * 8];
        v16h bf;
#pragma unroll
        for (int e = 0; e < 16; ++e) bf[e] = xr[(e & 7) + (e >> 3) * 16];
        acc[nt] = wmma_f16(a, bf, acc[nt]);
      }
    }
    // Epilogue: +bias, (Q: *1/sqrt(hd)), scatter to [win][head][token][hd].
#pragma unroll
    for (int nt = 0; nt < 4; ++nt) {
      const int p = nt * 16 + lm;  // pixel 0..63 -> token p+1
#pragma unroll
      for (int i = 0; i < 8; ++i) {
        const int m = mtile * 16 + hlf * 8 + i;  // inter channel 0..127
        float v = acc[nt][i] + bm[m];
        if (mat == 0) v *= QSCALE;
        const int o = ((w * NH + (m >> 3)) * NTOK + (p + 1)) * HD + (m & 7);
        Ows[o] = (_Float16)v;
      }
    }
  }
}

// ---------------------------------------------------------------------------
// Stage 2: attention. One block (4 waves) per window; each wave: 4 heads.
// S = Q K^T (WMMA, k padded 8->32), f32 softmax with shfl across 16 lanes,
// attn written to d_out (non-temporal: 553 MB write-once stream must not
// evict the L2-resident y tensor), P stashed in LDS f16, O = P V (WMMA).
// ---------------------------------------------------------------------------
__global__ __launch_bounds__(128) void k_attn(
    const _Float16* __restrict__ Qws, const _Float16* __restrict__ Kws,
    const _Float16* __restrict__ Vws, float* __restrict__ attn,
    _Float16* __restrict__ OUTws) {
  __shared__ __align__(16) _Float16 P[4][80][88];
  const int w = blockIdx.x;
  const int tid = threadIdx.x, wave = tid >> 5, lane = tid & 31;
  const int hlf = lane >> 4, lm = lane & 15;

  for (int hi = 0; hi < 4; ++hi) {
    const int h = wave * 4 + hi;
    const int base = (w * NH + h) * NTOK * HD;

    // K fragments (B operand of S): 5 column tiles; valid k only 0..7.
    v16h kb[5];
#pragma unroll
    for (int nt = 0; nt < 5; ++nt) {
      const int tk = nt * 16 + lm;
      _Float16 tmp[8];
      if (hlf == 0 && tk < NTOK) {
        const v8h r = *(const v8h*)(Kws + base + tk * HD);
#pragma unroll
        for (int e = 0; e < 8; ++e) tmp[e] = r[e];
      } else {
#pragma unroll
        for (int e = 0; e < 8; ++e) tmp[e] = (_Float16)0.0f;
      }
#pragma unroll
      for (int e = 0; e < 16; ++e)
        kb[nt][e] = (e < 8) ? tmp[e] : (_Float16)0.0f;
    }
    // V fragments (B operand of O): k = token (3 k-steps of 32), n = hd.
    v16h vb[3];
#pragma unroll
    for (int ks = 0; ks < 3; ++ks) {
#pragma unroll
      for (int e = 0; e < 16; ++e) {
        const int t = ks * 32 + hlf * 8 + (e & 7) + (e >> 3) * 16;
        _Float16 v = (_Float16)0.0f;
        if (t < NTOK && lm < HD) v = Vws[base + t * HD + lm];
        vb[ks][e] = v;
      }
    }

    for (int mt = 0; mt < 5; ++mt) {
      // Q fragment (A operand): row = query token, valid k only 0..7.
      v16h qa;
      {
        const int tq = mt * 16 + lm;
        _Float16 tmp[8];
        if (hlf == 0 && tq < NTOK) {
          const v8h r = *(const v8h*)(Qws + base + tq * HD);
#pragma unroll
          for (int e = 0; e < 8; ++e) tmp[e] = r[e];
        } else {
#pragma unroll
          for (int e = 0; e < 8; ++e) tmp[e] = (_Float16)0.0f;
        }
#pragma unroll
        for (int e = 0; e < 16; ++e)
          qa[e] = (e < 8) ? tmp[e] : (_Float16)0.0f;
      }

      v8f s[5];
#pragma unroll
      for (int nt = 0; nt < 5; ++nt) {
        v8f z = {};
        s[nt] = wmma_f16(qa, kb[nt], z);
      }
      // Mask key columns >= 65.
#pragma unroll
      for (int nt = 0; nt < 5; ++nt)
        if (nt * 16 + lm >= NTOK) {
#pragma unroll
          for (int i = 0; i < 8; ++i) s[nt][i] = -1e30f;
        }
      // Row softmax: row = mt*16 + hlf*8 + i lives across the 16 lanes lm.
      float rmax[8], rsum[8];
#pragma unroll
      for (int i = 0; i < 8; ++i) {
        float m = s[0][i];
#pragma unroll
        for (int nt = 1; nt < 5; ++nt) m = fmaxf(m, s[nt][i]);
#pragma unroll
        for (int d = 1; d < 16; d <<= 1) m = fmaxf(m, __shfl_xor(m, d, 32));
        rmax[i] = m;
        rsum[i] = 0.0f;
      }
#pragma unroll
      for (int nt = 0; nt < 5; ++nt) {
        const bool valid = (nt * 16 + lm) < NTOK;
#pragma unroll
        for (int i = 0; i < 8; ++i) {
          const float e = valid ? __expf(s[nt][i] - rmax[i]) : 0.0f;
          s[nt][i] = e;
          rsum[i] += e;
        }
      }
#pragma unroll
      for (int i = 0; i < 8; ++i) {
        float t = rsum[i];
#pragma unroll
        for (int d = 1; d < 16; d <<= 1) t += __shfl_xor(t, d, 32);
        rsum[i] = 1.0f / t;
      }
      // Normalize, emit attention (non-temporal), stash P strip in LDS.
#pragma unroll
      for (int nt = 0; nt < 5; ++nt) {
        const int n = nt * 16 + lm;
#pragma unroll
        for (int i = 0; i < 8; ++i) {
          const int m = mt * 16 + hlf * 8 + i;
          const float p = s[nt][i] * rsum[i];
          if (m < NTOK && n < NTOK)
            __builtin_nontemporal_store(
                p, &attn[((w * NH + h) * NTOK + m) * NTOK + n]);
          P[wave][m][n] = (_Float16)p;
        }
      }
      // O strip = P strip x V  (k padded 80 -> 96; pad cols of P are zero in vb)
      v8f o = {};
#pragma unroll
      for (int ks = 0; ks < 3; ++ks) {
        v16h pa;
        const int row = mt * 16 + lm;
#pragma unroll
        for (int g = 0; g < 2; ++g) {
          const int k0 = ks * 32 + g * 16 + hlf * 8;
          if (k0 < 80) {
            const v8h r = *(const v8h*)&P[wave][row][k0];
#pragma unroll
            for (int e = 0; e < 8; ++e) pa[g * 8 + e] = r[e];
          } else {
#pragma unroll
            for (int e = 0; e < 8; ++e) pa[g * 8 + e] = (_Float16)0.0f;
          }
        }
        o = wmma_f16(pa, vb[ks], o);
      }
      // Drop global-token row; store [win][inter_channel][pixel] as f16.
      if (lm < HD) {
        const int ic = h * HD + lm;
#pragma unroll
        for (int i = 0; i < 8; ++i) {
          const int m = mt * 16 + hlf * 8 + i;
          if (m >= 1 && m < NTOK)
            OUTws[(w * INTER + ic) * 64 + (m - 1)] = (_Float16)o[i];
        }
      }
    }
  }
}

// ---------------------------------------------------------------------------
// Stage 3: z-projection. One block per window. GEMM M=256,N=64,K=128.
// Writes pre-BN y into the z region of d_out (regular-temporal: re-read by
// the reduce and BN kernels; 134 MB fits in the 192 MB L2).
// ---------------------------------------------------------------------------
__global__ __launch_bounds__(256) void k_zproj(
    const _Float16* __restrict__ OUTws, const float* __restrict__ Wz,
    const float* __restrict__ bz, float* __restrict__ y) {
  __shared__ __align__(16) _Float16 WzL[CIN][136];
  __shared__ __align__(16) _Float16 OT[64][136];
  const int w = blockIdx.x;
  const int b = w >> 8, wh = (w >> 4) & 15, ww = w & 15;
  const int tid = threadIdx.x;

  for (int idx = tid; idx < CIN * INTER; idx += 256) {
    WzL[idx >> 7][idx & 127] = (_Float16)Wz[idx];
  }
  for (int idx = tid; idx < INTER * 64; idx += 256) {
    const int ic = idx >> 6, t = idx & 63;
    OT[t][ic] = OUTws[(w * INTER + ic) * 64 + t];
  }
  __syncthreads();

  const int lane = tid & 31, hlf = lane >> 4, lm = lane & 15;
  const int wave = tid >> 5;  // wave handles M tiles {wave, wave+8}

  v8f acc[2][4] = {};
  for (int ks = 0; ks < 4; ++ks) {
    v16h a[2];
#pragma unroll
    for (int mt2 = 0; mt2 < 2; ++mt2) {
      const int m = (wave + mt2 * 8) * 16 + lm;
      const _Float16* wr = &WzL[m][ks * 32 + hlf * 8];
#pragma unroll
      for (int g = 0; g < 2; ++g)
#pragma unroll
        for (int e = 0; e < 8; ++e) a[mt2][g * 8 + e] = wr[g * 16 + e];
    }
#pragma unroll
    for (int nt = 0; nt < 4; ++nt) {
      const _Float16* xr = &OT[nt * 16 + lm][ks * 32 + hlf * 8];
      v16h bf;
#pragma unroll
      for (int e = 0; e < 16; ++e) bf[e] = xr[(e & 7) + (e >> 3) * 16];
#pragma unroll
      for (int mt2 = 0; mt2 < 2; ++mt2)
        acc[mt2][nt] = wmma_f16(a[mt2], bf, acc[mt2][nt]);
    }
  }
#pragma unroll
  for (int mt2 = 0; mt2 < 2; ++mt2) {
#pragma unroll
    for (int nt = 0; nt < 4; ++nt) {
      const int p = nt * 16 + lm;
      const int hp = wh * 8 + (p >> 3), wp = ww * 8 + (p & 7);
#pragma unroll
      for (int i = 0; i < 8; ++i) {
        const int oc = (wave + mt2 * 8) * 16 + hlf * 8 + i;
        y[((b * CIN + oc) * IMG + hp) * IMG + wp] = acc[mt2][nt][i] + bz[oc];
      }
    }
  }
}

// ---------------------------------------------------------------------------
// Stage 4: per-channel sum / sum-of-squares (one block per channel).
// ---------------------------------------------------------------------------
__global__ __launch_bounds__(256) void k_reduce(const float* __restrict__ y,
                                                float* __restrict__ sums) {
  const int c = blockIdx.x, tid = threadIdx.x;
  float s = 0.0f, s2 = 0.0f;
  for (int b = 0; b < 8; ++b) {
    const float* p = y + ((size_t)b * CIN + c) * (IMG * IMG);
    for (int i = tid; i < IMG * IMG; i += 256) {
      const float v = p[i];
      s += v;
      s2 += v * v;
    }
  }
  __shared__ float rs[256], rs2[256];
  rs[tid] = s;
  rs2[tid] = s2;
  __syncthreads();
  for (int off = 128; off > 0; off >>= 1) {
    if (tid < off) {
      rs[tid] += rs[tid + off];
      rs2[tid] += rs2[tid + off];
    }
    __syncthreads();
  }
  if (tid == 0) {
    sums[c] = rs[0];
    sums[CIN + c] = rs2[0];
  }
}

// ---------------------------------------------------------------------------
// Stage 5: BatchNorm (batch stats) + gamma/beta + residual, in place.
// Final z is write-once: non-temporal store keeps it from churning L2.
// ---------------------------------------------------------------------------
__global__ __launch_bounds__(256) void k_bn(float* __restrict__ y,
                                            const float* __restrict__ xthis,
                                            const float* __restrict__ sums,
                                            const float* __restrict__ gamma,
                                            const float* __restrict__ beta) {
  const int idx = blockIdx.x * 256 + threadIdx.x;
  const int c = (idx >> 14) & (CIN - 1);
  const float invn = 1.0f / (8.0f * IMG * IMG);
  const float mean = sums[c] * invn;
  const float var = sums[CIN + c] * invn - mean * mean;
  const float sc = rsqrtf(var + 1e-5f) * gamma[c];
  const float out = (y[idx] - mean) * sc + beta[c] + xthis[idx];
  __builtin_nontemporal_store(out, &y[idx]);
}

// ---------------------------------------------------------------------------
extern "C" void kernel_launch(void* const* d_in, const int* in_sizes, int n_in,
                              void* d_out, int out_size, void* d_ws,
                              size_t ws_size, hipStream_t stream) {
  const float* xthis  = (const float*)d_in[0];
  const float* xother = (const float*)d_in[1];
  const float* Wq = (const float*)d_in[2];
  const float* bq = (const float*)d_in[3];
  const float* Wk = (const float*)d_in[4];
  const float* bk = (const float*)d_in[5];
  const float* Wv = (const float*)d_in[6];
  const float* bv = (const float*)d_in[7];
  const float* gtok = (const float*)d_in[8];
  const float* Wz = (const float*)d_in[9];
  const float* bz = (const float*)d_in[10];
  const float* gamma = (const float*)d_in[11];
  const float* beta = (const float*)d_in[12];

  const size_t QKV = (size_t)NWIN * NH * NTOK * HD;  // f16 elements each
  _Float16* Qws = (_Float16*)d_ws;
  _Float16* Kws = Qws + QKV;
  _Float16* Vws = Kws + QKV;
  _Float16* OUTws = Vws + QKV;
  float* sums = (float*)(OUTws + (size_t)NWIN * INTER * 64);

  float* z = (float*)d_out;                       // [8,256,128,128]
  float* attn = z + (size_t)8 * CIN * IMG * IMG;  // [2048,16,65,65]

  k_qkv<<<NWIN, 256, 0, stream>>>(xthis, xother, Wq, bq, Wk, bk, Wv, bv, gtok,
                                  Qws, Kws, Vws);
  k_attn<<<NWIN, 128, 0, stream>>>(Qws, Kws, Vws, attn, OUTws);
  k_zproj<<<NWIN, 256, 0, stream>>>(OUTws, Wz, bz, z);
  k_reduce<<<CIN, 256, 0, stream>>>(z, sums);
  k_bn<<<(8 * CIN * IMG * IMG) / 256, 256, 0, stream>>>(z, xthis, sums, gamma,
                                                        beta);
}
